// FeedForwardSpeakerDecoder_84069689852280
// MI455X (gfx1250) — matrix operand order repaired and checked by reading, not compiled, not run on hardware
//
#include <hip/hip_runtime.h>
#include <hip/hip_bf16.h>

// Sizes from the reference
#define VOCAB 32000
#define EMB   256
#define HID   512
#define BB    32
#define TT    80
#define SS    20
#define G4    2048   // 4*HID

typedef __attribute__((ext_vector_type(16))) __bf16 v16bf;
typedef __attribute__((ext_vector_type(8)))  float  v8f;
typedef __attribute__((ext_vector_type(4)))  unsigned int u32x4;
typedef __attribute__((ext_vector_type(8)))  int    i32x8;
typedef __attribute__((ext_vector_type(4)))  int    i32x4;
typedef __hip_bfloat16 bf16_t;

#if defined(__has_builtin)
#if __has_builtin(__builtin_amdgcn_tensor_load_to_lds)
#define USE_TDM 1
#endif
#endif
#ifndef USE_TDM
#define USE_TDM 0
#endif

// ---------------------------------------------------------------------------
// WMMA helpers: D = A(16x32 bf16) x B(32x16 bf16) + C(16x16 f32)
// ---------------------------------------------------------------------------
__device__ __forceinline__ v8f wmma_bf16(v16bf a, v16bf b, v8f c) {
  return __builtin_amdgcn_wmma_f32_16x16x32_bf16(
      /*neg_a=*/false, a, /*neg_b=*/false, b,
      /*c_mod=*/(short)0, c, /*reuse_a=*/false, /*reuse_b=*/false);
}

// A tile (16x32, row-major source, lda elements):
// lanes 0-15 : rows m0+lane,   elems 0..7 = K k0..k0+7,  elems 8..15 = K k0+16..k0+23
// lanes 16-31: rows m0+lane-16,elems 0..7 = K k0+8..+15, elems 8..15 = K k0+24..+31
__device__ __forceinline__ v16bf load_a_bf16(const bf16_t* A, int lda,
                                             int m0, int k0, int lane) {
  int row = m0 + (lane & 15);
  int kc  = k0 + ((lane & 16) ? 8 : 0);
  const bf16_t* p = A + (size_t)row * lda + kc;
  v16bf r;
  ((uint4*)&r)[0] = *(const uint4*)(p);
  ((uint4*)&r)[1] = *(const uint4*)(p + 16);
  return r;
}

// B tile (32x16) from weight matrix stored [N x K] row-major (W[n][k]):
// lane n (0-15): K = k0..k0+7, k0+16..k0+23; lane n+16: K = k0+8..k0+15, k0+24..k0+31
__device__ __forceinline__ v16bf load_b_bf16(const bf16_t* W, int ldb,
                                             int n0, int k0, int lane) {
  int col = n0 + (lane & 15);
  int kc  = k0 + ((lane & 16) ? 8 : 0);
  const bf16_t* p = W + (size_t)col * ldb + kc;
  v16bf r;
  ((uint4*)&r)[0] = *(const uint4*)(p);
  ((uint4*)&r)[1] = *(const uint4*)(p + 16);
  return r;
}

__device__ __forceinline__ float sigf(float x) {
  return 1.0f / (1.0f + __expf(-x));
}

// ---------------------------------------------------------------------------
// f32 -> bf16 conversion (weights, done once per launch)
// ---------------------------------------------------------------------------
__global__ void cvt_f32_bf16(const float* __restrict__ in,
                             bf16_t* __restrict__ out, int n) {
  int i = blockIdx.x * blockDim.x + threadIdx.x;
  int stride = gridDim.x * blockDim.x;
  for (; i < n; i += stride) out[i] = __float2bfloat16(in[i]);
}

// Embedding gather -> bf16, rows m = t*B + b (time-major like the scan)
__global__ void embed_gather(const int* __restrict__ words,
                             const float* __restrict__ emb,
                             bf16_t* __restrict__ xbf) {
  int idx = blockIdx.x * 256 + threadIdx.x;        // < 2560*256
  int m = idx >> 8, k = idx & 255;
  int t = m / BB, b = m % BB;
  int w = words[b * TT + t];
  xbf[idx] = __float2bfloat16(emb[(size_t)w * EMB + k]);
}

// ---------------------------------------------------------------------------
// Input projection, hoisted out of the time loop:
// gates_x[2560,2048] = xbf[2560,256] @ W_ih^T + (b_ih + b_hh)
// Block: 256 thr = 8 waves, tile 16M x 512N, A tile staged in LDS.
// ---------------------------------------------------------------------------
__global__ void gates_gemm(const bf16_t* __restrict__ A,
                           const bf16_t* __restrict__ W,     // [2048,256]
                           const float* __restrict__ bih,
                           const float* __restrict__ bhh,
                           float* __restrict__ C) {
  extern __shared__ char smem[];
  bf16_t* sA = (bf16_t*)smem;                       // 16 x 256
  const int m0   = blockIdx.y * 16;
  const int lane = threadIdx.x & 31;
  const int wave = threadIdx.x >> 5;
  {
    const uint4* src = (const uint4*)(A + (size_t)m0 * EMB);
    uint4* dst = (uint4*)sA;
    for (int i = threadIdx.x; i < (16 * EMB) / 8; i += blockDim.x) dst[i] = src[i];
  }
  __syncthreads();
  const int nbase = blockIdx.x * 512 + wave * 64;
  v8f acc[4];
#pragma unroll
  for (int j = 0; j < 4; ++j) acc[j] = (v8f){0, 0, 0, 0, 0, 0, 0, 0};
  for (int k0 = 0; k0 < EMB; k0 += 32) {
    v16bf a = load_a_bf16(sA, EMB, 0, k0, lane);
#pragma unroll
    for (int j = 0; j < 4; ++j) {
      v16bf b = load_b_bf16(W, EMB, nbase + j * 16, k0, lane);
      acc[j] = wmma_bf16(a, b, acc[j]);
    }
  }
  const int rowoff = (lane & 16) ? 8 : 0;
#pragma unroll
  for (int j = 0; j < 4; ++j) {
    int n = nbase + j * 16 + (lane & 15);
    float bias = bih[n] + bhh[n];
#pragma unroll
    for (int r = 0; r < 8; ++r) {
      int m = m0 + r + rowoff;
      C[(size_t)m * G4 + n] = acc[j][r] + bias;
    }
  }
}

// ---------------------------------------------------------------------------
// Recurrent kernel: one workgroup (512 thr = 16 waves) loops all T=80 steps.
// LDS (~200KB of the 320KB WGP LDS): h(bf16), h1(bf16), c(f32),
// target(f32)/wctx(bf16 overlay), attention weights.
// ---------------------------------------------------------------------------
__global__ __launch_bounds__(512) void lstm_attn_recurrent(
    const float* __restrict__ gx,        // gates_x [2560,2048]
    const bf16_t* __restrict__ Whh,      // [2048,512] bf16
    const bf16_t* __restrict__ Win,      // [512,512]  bf16
    const bf16_t* __restrict__ Wout,     // [512,1024] bf16
    const float* __restrict__ ctx,       // [32,20,512]
    const unsigned char* __restrict__ mask,  // [32,20] bool
    const float* __restrict__ h0,
    const float* __restrict__ c0,
    bf16_t* __restrict__ htilde_bf,      // [2560,512] bf16 (rows m = t*B+b)
    float* __restrict__ hN,              // [32,512]
    float* __restrict__ cN) {            // [32,512]
  extern __shared__ char smem[];
  bf16_t* sh_h  = (bf16_t*)smem;                 // 32*512 bf16 = 32768B
  bf16_t* sh_h1 = (bf16_t*)(smem + 32768);       // 32768B
  float*  sh_c  = (float*)(smem + 65536);        // 65536B
  float*  sh_t  = (float*)(smem + 131072);       // 65536B (target)
  bf16_t* sh_w  = (bf16_t*)(smem + 131072);      // wctx overlays target
  float*  sh_aw = (float*)(smem + 196608);       // 32*20*4 = 2560B

  const int tid  = threadIdx.x;
  const int lane = tid & 31;
  const int wave = tid >> 5;   // 0..15
  const v8f vz = (v8f){0, 0, 0, 0, 0, 0, 0, 0};

  for (int i = tid; i < BB * HID; i += 512) {
    sh_h[i] = __float2bfloat16(h0[i]);
    sh_c[i] = c0[i];
  }
  __syncthreads();

  for (int t = 0; t < TT; ++t) {
    // ---- Phase 1: gates = gates_x[t] + h @ W_hh^T ; LSTM pointwise ----
    for (int chunk = wave; chunk < HID / 16; chunk += 16) {
      int n0 = chunk * 16;
      v8f acc[4][2];
#pragma unroll
      for (int g = 0; g < 4; ++g) { acc[g][0] = vz; acc[g][1] = vz; }
      for (int k0 = 0; k0 < HID; k0 += 32) {
        v16bf a0 = load_a_bf16(sh_h, HID, 0, k0, lane);
        v16bf a1 = load_a_bf16(sh_h, HID, 16, k0, lane);
#pragma unroll
        for (int g = 0; g < 4; ++g) {
          v16bf b = load_b_bf16(Whh, HID, g * HID + n0, k0, lane);
          acc[g][0] = wmma_bf16(a0, b, acc[g][0]);
          acc[g][1] = wmma_bf16(a1, b, acc[g][1]);
        }
      }
      const int rowoff = (lane & 16) ? 8 : 0;
      const int n = n0 + (lane & 15);
#pragma unroll
      for (int mt = 0; mt < 2; ++mt) {
#pragma unroll
        for (int r = 0; r < 8; ++r) {
          int b = mt * 16 + r + rowoff;
          const float* grow = gx + (size_t)(t * BB + b) * G4;
          float gi = acc[0][mt][r] + grow[n];
          float gf = acc[1][mt][r] + grow[HID + n];
          float gg = acc[2][mt][r] + grow[2 * HID + n];
          float go = acc[3][mt][r] + grow[3 * HID + n];
          float i_ = sigf(gi), f_ = sigf(gf), o_ = sigf(go);
          float g_ = tanhf(gg);
          float c1 = f_ * sh_c[b * HID + n] + i_ * g_;
          float h1 = o_ * tanhf(c1);
          sh_c[b * HID + n]  = c1;
          sh_h1[b * HID + n] = __float2bfloat16(h1);
        }
      }
    }
    __syncthreads();

    // ---- Phase 2: target = h1 @ W_in^T ----
    for (int chunk = wave; chunk < HID / 16; chunk += 16) {
      int n0 = chunk * 16;
      v8f a0c = vz, a1c = vz;
      for (int k0 = 0; k0 < HID; k0 += 32) {
        v16bf a0 = load_a_bf16(sh_h1, HID, 0, k0, lane);
        v16bf a1 = load_a_bf16(sh_h1, HID, 16, k0, lane);
        v16bf b  = load_b_bf16(Win, HID, n0, k0, lane);
        a0c = wmma_bf16(a0, b, a0c);
        a1c = wmma_bf16(a1, b, a1c);
      }
      const int rowoff = (lane & 16) ? 8 : 0;
      const int n = n0 + (lane & 15);
#pragma unroll
      for (int r = 0; r < 8; ++r) {
        sh_t[(r + rowoff) * HID + n]      = a0c[r];
        sh_t[(16 + r + rowoff) * HID + n] = a1c[r];
      }
    }
    __syncthreads();

    // ---- Phase 3: attn scores ----
    for (int p = tid; p < BB * SS; p += 512) {
      int b = p / SS, s = p % SS;
      const float* crow = ctx + (size_t)(b * SS + s) * HID;
      float acc = 0.f;
      for (int d = 0; d < HID; ++d) acc += crow[d] * sh_t[b * HID + d];
      if (mask[p]) acc = -__builtin_inff();
      sh_aw[p] = acc;
    }
    __syncthreads();

    // ---- Phase 4: softmax over S per batch row ----
    if (tid < BB) {
      float mx = -__builtin_inff();
      for (int s = 0; s < SS; ++s) mx = fmaxf(mx, sh_aw[tid * SS + s]);
      float sum = 0.f;
      for (int s = 0; s < SS; ++s) {
        float e = __expf(sh_aw[tid * SS + s] - mx);
        sh_aw[tid * SS + s] = e;
        sum += e;
      }
      float inv = 1.f / sum;
      for (int s = 0; s < SS; ++s) sh_aw[tid * SS + s] *= inv;
    }
    __syncthreads();

    // ---- Phase 5: wctx = a @ ctx  (overlay onto target region, bf16) ----
    for (int idx = tid; idx < BB * HID; idx += 512) {
      int b = idx >> 9, d = idx & (HID - 1);
      float acc = 0.f;
#pragma unroll
      for (int s = 0; s < SS; ++s)
        acc += sh_aw[b * SS + s] * ctx[(size_t)(b * SS + s) * HID + d];
      sh_w[idx] = __float2bfloat16(acc);
    }
    __syncthreads();

    // ---- Phase 6: h_tilde = tanh([wctx, h1] @ W_out^T) ----
    for (int chunk = wave; chunk < HID / 16; chunk += 16) {
      int n0 = chunk * 16;
      v8f a0c = vz, a1c = vz;
      for (int k0 = 0; k0 < 2 * HID; k0 += 32) {
        const bf16_t* asrc = (k0 < HID) ? sh_w : sh_h1;
        int kk = k0 & (HID - 1);
        v16bf a0 = load_a_bf16(asrc, HID, 0, kk, lane);
        v16bf a1 = load_a_bf16(asrc, HID, 16, kk, lane);
        v16bf b  = load_b_bf16(Wout, 2 * HID, n0, k0, lane);
        a0c = wmma_bf16(a0, b, a0c);
        a1c = wmma_bf16(a1, b, a1c);
      }
      const int rowoff = (lane & 16) ? 8 : 0;
      const int n = n0 + (lane & 15);
#pragma unroll
      for (int mt = 0; mt < 2; ++mt) {
        v8f sel = mt ? a1c : a0c;
#pragma unroll
        for (int r = 0; r < 8; ++r) {
          int b = mt * 16 + r + rowoff;
          float ht = tanhf(sel[r]);
          bf16_t hb = __float2bfloat16(ht);
          sh_h[b * HID + n] = hb;                              // next h_prev
          htilde_bf[(size_t)(t * BB + b) * HID + n] = hb;      // GEMM input
          if (t == TT - 1) hN[b * HID + n] = ht;
        }
      }
    }
    __syncthreads();
  }

  for (int i = tid; i < BB * HID; i += 512) cN[i] = sh_c[i];
}

// ---------------------------------------------------------------------------
// Vocab projection, batched over all (t,b):
// logits = htilde_bf[2560,512] @ W_proj^T + b_proj -> out[b][t][v]
// Grid (63, 80); block 256 thr = 8 waves; block tile 32M x 512N.
// Each wave owns 2 M-tiles x 4 N-tiles so every B tile feeds 2 WMMAs.
// A tile (32x512 bf16 = 32KB, contiguous since lda==K) is staged into LDS by
// the Tensor Data Mover (tensor_load_to_lds + s_wait_tensorcnt) when the
// toolchain exposes it; plain cooperative copy otherwise.
// ---------------------------------------------------------------------------
__global__ void proj_gemm(const bf16_t* __restrict__ A,
                          const bf16_t* __restrict__ W,     // [32000,512]
                          const float* __restrict__ bias,
                          float* __restrict__ out) {
  extern __shared__ char smem[];
  bf16_t* sA = (bf16_t*)smem;                       // 32 x 512
  const int m0   = blockIdx.y * 32;
  const int lane = threadIdx.x & 31;
  const int wave = threadIdx.x >> 5;

#if USE_TDM
  if (wave == 0) {
    // D# group 0: count=1, lds_addr=0 (dynamic-LDS base), global byte addr,
    // type=2 ("image") in bits [127:126].
    unsigned long long ga = (unsigned long long)(const void*)(A + (size_t)m0 * HID);
    u32x4 g0;
    g0[0] = 1u;                                       // count=1, user D#
    g0[1] = 0u;                                       // lds_addr
    g0[2] = (unsigned)(ga & 0xFFFFFFFFu);             // global_addr[31:0]
    g0[3] = (unsigned)((ga >> 32) & 0x01FFFFFFu) | (2u << 30);
    // D# group 1: data_size=2B, tensor_dim0=tile_dim0=16384 elems (one row),
    // tile_dim1=1, tensor_dim0_stride=16384.
    i32x8 g1;
    g1[0] = (int)(1u << 16);                          // data_size=1 -> 2 bytes
    g1[1] = (int)(0x4000u << 16);                     // tensor_dim0[15:0]=16384
    g1[2] = (int)(1u << 16);                          // dim0 hi=0, tensor_dim1=1
    g1[3] = (int)(0x4000u << 16);                     // dim1 hi=0, tile_dim0=16384
    g1[4] = 1;                                        // tile_dim1=1, tile_dim2=0
    g1[5] = 0x4000;                                   // tensor_dim0_stride=16384
    g1[6] = (int)(0x4000u << 16);                     // stride0 hi=0, dim1_stride lo
    g1[7] = 0;
    i32x4 z4 = (i32x4){0, 0, 0, 0};
#if __clang_major__ >= 23
    i32x8 z8 = (i32x8){0, 0, 0, 0, 0, 0, 0, 0};
    __builtin_amdgcn_tensor_load_to_lds(g0, g1, z4, z4, z8, 0);
#else
    __builtin_amdgcn_tensor_load_to_lds(g0, g1, z4, z4, 0);
#endif
#if __has_builtin(__builtin_amdgcn_s_wait_tensorcnt)
    __builtin_amdgcn_s_wait_tensorcnt(0);
#else
    asm volatile("s_wait_tensorcnt 0x0" ::: "memory");
#endif
  }
#else
  {
    const uint4* src = (const uint4*)(A + (size_t)m0 * HID);
    uint4* dst = (uint4*)sA;
    for (int i = threadIdx.x; i < (32 * HID) / 8; i += blockDim.x) dst[i] = src[i];
  }
#endif
  __syncthreads();

  const int nbase = blockIdx.x * 512 + wave * 64;
  v8f acc[2][4];
#pragma unroll
  for (int mt = 0; mt < 2; ++mt)
#pragma unroll
    for (int j = 0; j < 4; ++j) acc[mt][j] = (v8f){0, 0, 0, 0, 0, 0, 0, 0};

  for (int k0 = 0; k0 < HID; k0 += 32) {
    v16bf a0 = load_a_bf16(sA, HID, 0, k0, lane);
    v16bf a1 = load_a_bf16(sA, HID, 16, k0, lane);
    if (nbase < VOCAB)
      __builtin_prefetch((const char*)(W + (size_t)(nbase + (lane & 15)) * HID + k0) + 256, 0, 1);
#pragma unroll
    for (int j = 0; j < 4; ++j) {
      int nt = nbase + j * 16;
      if (nt < VOCAB) {                               // wave-uniform guard
        v16bf b = load_b_bf16(W, HID, nt, k0, lane);
        acc[0][j] = wmma_bf16(a0, b, acc[0][j]);
        acc[1][j] = wmma_bf16(a1, b, acc[1][j]);
      }
    }
  }

  const int rowoff = (lane & 16) ? 8 : 0;
#pragma unroll
  for (int j = 0; j < 4; ++j) {
    int nt = nbase + j * 16;
    if (nt < VOCAB) {
      int v = nt + (lane & 15);
      float bv = bias[v];
#pragma unroll
      for (int mt = 0; mt < 2; ++mt) {
#pragma unroll
        for (int r = 0; r < 8; ++r) {
          int m = m0 + mt * 16 + r + rowoff;
          int b = m & 31, t = m >> 5;
          out[((size_t)b * TT + t) * VOCAB + v] = acc[mt][j][r] + bv;
        }
      }
    }
  }
}

// ---------------------------------------------------------------------------
// Launch
// ---------------------------------------------------------------------------
extern "C" void kernel_launch(void* const* d_in, const int* in_sizes, int n_in,
                              void* d_out, int out_size, void* d_ws, size_t ws_size,
                              hipStream_t stream) {
  (void)in_sizes; (void)n_in; (void)out_size; (void)ws_size;

  const int*           words  = (const int*)d_in[0];
  const float*         ctx    = (const float*)d_in[1];
  const unsigned char* mask   = (const unsigned char*)d_in[2];
  const float*         h0     = (const float*)d_in[3];
  const float*         c0     = (const float*)d_in[4];
  const float*         emb    = (const float*)d_in[5];
  const float*         W_ih   = (const float*)d_in[6];
  const float*         W_hh   = (const float*)d_in[7];
  const float*         b_ih   = (const float*)d_in[8];
  const float*         b_hh   = (const float*)d_in[9];
  const float*         W_in   = (const float*)d_in[10];
  const float*         W_out  = (const float*)d_in[11];
  const float*         W_proj = (const float*)d_in[12];
  const float*         b_proj = (const float*)d_in[13];

  char* ws = (char*)d_ws;                          // static carve (~62.4 MB)
  bf16_t* xbf    = (bf16_t*)(ws + 0);              // 2560*256  bf16
  bf16_t* Wih_b  = (bf16_t*)(ws + 1310720);        // 2048*256  bf16
  bf16_t* Whh_b  = (bf16_t*)(ws + 2359296);        // 2048*512  bf16
  bf16_t* Win_b  = (bf16_t*)(ws + 4456448);        // 512*512   bf16
  bf16_t* Wout_b = (bf16_t*)(ws + 4980736);        // 512*1024  bf16
  bf16_t* Wprj_b = (bf16_t*)(ws + 6029312);        // 32000*512 bf16
  float*  gx     = (float*)(ws + 38797312);        // 2560*2048 f32
  bf16_t* htb    = (bf16_t*)(ws + 59768832);       // 2560*512  bf16

  float* logits = (float*)d_out;                   // [32,80,32000]
  float* hN = logits + (size_t)BB * TT * VOCAB;    // [1,32,512]
  float* cN = hN + BB * HID;                       // [1,32,512]

  cvt_f32_bf16<<<(2048 * 256 + 255) / 256, 256, 0, stream>>>(W_ih, Wih_b, 2048 * 256);
  cvt_f32_bf16<<<(2048 * 512 + 255) / 256, 256, 0, stream>>>(W_hh, Whh_b, 2048 * 512);
  cvt_f32_bf16<<<(512 * 512 + 255) / 256, 256, 0, stream>>>(W_in, Win_b, 512 * 512);
  cvt_f32_bf16<<<(512 * 1024 + 255) / 256, 256, 0, stream>>>(W_out, Wout_b, 512 * 1024);
  cvt_f32_bf16<<<(VOCAB * 512 + 255) / 256, 256, 0, stream>>>(W_proj, Wprj_b, VOCAB * 512);

  embed_gather<<<2560, 256, 0, stream>>>(words, emb, xbf);

  gates_gemm<<<dim3(4, 160), 256, 16 * EMB * 2, stream>>>(xbf, Wih_b, b_ih, b_hh, gx);

  lstm_attn_recurrent<<<1, 512, 199168, stream>>>(gx, Whh_b, Win_b, Wout_b,
                                                  ctx, mask, h0, c0, htb, hN, cN);

  proj_gemm<<<dim3(63, 80), 256, 32 * HID * 2, stream>>>(htb, Wprj_b, b_proj, logits);
}
